// ConvHex_3762391351449
// MI455X (gfx1250) — matrix operand order repaired and checked by reading, not compiled, not run on hardware
//
#include <hip/hip_runtime.h>

typedef _Float16 f16;
typedef _Float16 v16h __attribute__((ext_vector_type(16)));
typedef _Float16 v8h  __attribute__((ext_vector_type(8)));
typedef float    v8f  __attribute__((ext_vector_type(8)));

#define B_     32
#define CIN    32
#define COUT   64
#define NPTS   20000
#define ROWS   (NPTS + 1)     // +1 zero row per batch for invalid neighbors
#define KN     6
#define NCHUNK 7              // 1 center + 6 neighbors; K = 7*32 = 224
#define NT32   (NPTS / 32)    // 625 blocks of 32 columns per batch

// d_ws layout: [0, 32KB) packed f16 weights Wp[COUT][NCHUNK*CIN]
//              [32KB, 32KB + B*ROWS*CIN*2) transposed f16 features xt[B][ROWS][CIN]
#define XT_OFF (32 * 1024)

// ---------------------------------------------------------------------------
// Wp[o][j*32 + c] = j==0 ? Wc[o][c] : Wn[o][c][j-1]   (f16)
__global__ void pack_weights(const float* __restrict__ wc,
                             const float* __restrict__ wn,
                             f16* __restrict__ wp) {
  int t = blockIdx.x * blockDim.x + threadIdx.x;
  const int total = COUT * NCHUNK * CIN;
  if (t >= total) return;
  int c = t & (CIN - 1);
  int j = (t / CIN) % NCHUNK;
  int o = t / (CIN * NCHUNK);
  float v = (j == 0) ? wc[o * CIN + c] : wn[(o * CIN + c) * KN + (j - 1)];
  wp[o * (NCHUNK * CIN) + j * CIN + c] = (f16)v;
}

// ---------------------------------------------------------------------------
// xt[b][n][c] = (f16) x[b][c][n]; row n == NPTS is all zeros (invalid target).
__global__ void transpose_x(const float* __restrict__ x, f16* __restrict__ xt) {
  int t = blockIdx.x * blockDim.x + threadIdx.x;  // (b,row) flat over B*ROWS
  if (t >= B_ * ROWS) return;
  int b = t / ROWS;
  int n = t - b * ROWS;
  v8h out[4] = {};
  if (n < NPTS) {
    const float* src = x + (size_t)b * CIN * NPTS + n;
#pragma unroll
    for (int c8 = 0; c8 < 4; ++c8) {
      v8h o;
#pragma unroll
      for (int i = 0; i < 8; ++i) o[i] = (f16)src[(size_t)(c8 * 8 + i) * NPTS];
      out[c8] = o;
    }
  }
  v8h* dst = (v8h*)(xt + (size_t)t * CIN);
#pragma unroll
  for (int c8 = 0; c8 < 4; ++c8) dst[c8] = out[c8];
}

// ---------------------------------------------------------------------------
// Block = (b, 32-node tile), 4 waves; wave w = output rows [16w, 16w+16) and
// covers two 16-column tiles, reusing the resident A (weight) fragments.
__global__ __launch_bounds__(128) void convhex_wmma(
    const f16* __restrict__ xt, const f16* __restrict__ wp,
    const int* __restrict__ nbrs, const float* __restrict__ bias,
    float* __restrict__ out) {
  const int lane = threadIdx.x & 31;
  const int wave = threadIdx.x >> 5;        // o-tile 0..3
  const int blk  = blockIdx.x;              // b * NT32 + nt
  const int b    = blk / NT32;
  const int n0   = (blk - b * NT32) * 32;   // first of 32 columns

  const int col = lane & 15;                // B/D column within a 16-tile
  const int hi  = lane >> 4;                // 0: low K half, 1: high K half

  // ---- A fragments (weights), resident in VGPRs for all 7 chunks ----
  // lanes 0-15: elems 0..15 = K[0..16)
  // lanes16-31: elems 0..15 = K[16..32)
  const int cA = hi * 16;
  const int orow = wave * 16 + col;
  v16h a[NCHUNK];
#pragma unroll
  for (int j = 0; j < NCHUNK; ++j) {
    const f16* wr = wp + orow * (NCHUNK * CIN) + j * CIN + cA;
    union { v16h v; v8h h[2]; } u;
    u.h[0] = *(const v8h*)(wr);
    u.h[1] = *(const v8h*)(wr + 8);
    a[j] = u.v;
  }

  // ---- per-lane gather byte-offsets; invalid -> zero row (ROWS-1) ----
  // B layout: lane holds 16 contiguous K (channels): lanes 0-15 -> c[0..16),
  // lanes 16-31 -> c[16..32)  => 32 contiguous bytes per lane per chunk.
  const unsigned cOffB = (unsigned)(hi * 16) * sizeof(f16);
  unsigned soff0[NCHUNK], soff1[NCHUNK];
  {
    const int c0 = n0 + col;        // tile 0 column
    const int c1 = n0 + 16 + col;   // tile 1 column
    soff0[0] = (unsigned)c0 * (CIN * sizeof(f16)) + cOffB;
    soff1[0] = (unsigned)c1 * (CIN * sizeof(f16)) + cOffB;
#pragma unroll
    for (int k = 0; k < KN; ++k) {
      int s0 = nbrs[c0 * KN + k];
      int s1 = nbrs[c1 * KN + k];
      unsigned r0 = (s0 >= 0) ? (unsigned)s0 : (unsigned)NPTS;  // zero row
      unsigned r1 = (s1 >= 0) ? (unsigned)s1 : (unsigned)NPTS;
      soff0[1 + k] = r0 * (CIN * sizeof(f16)) + cOffB;
      soff1[1 + k] = r1 * (CIN * sizeof(f16)) + cOffB;
    }
  }

  // ---- 14 chained v_wmma_f32_16x16x32_f16 (7 per column tile) ----
  const char* xb = (const char*)(xt + (size_t)b * ROWS * CIN);  // uniform base
  v8f acc0 = {}, acc1 = {};
#pragma unroll
  for (int j = 0; j < NCHUNK; ++j) {
    union { v16h v; v8h h[2]; } u;
    u.h[0] = *(const v8h*)(xb + soff0[j]);
    u.h[1] = *(const v8h*)(xb + soff0[j] + 16);
    acc0 = __builtin_amdgcn_wmma_f32_16x16x32_f16(
        false, a[j], false, u.v, (short)0, acc0, false, false);
  }
#pragma unroll
  for (int j = 0; j < NCHUNK; ++j) {
    union { v16h v; v8h h[2]; } u;
    u.h[0] = *(const v8h*)(xb + soff1[j]);
    u.h[1] = *(const v8h*)(xb + soff1[j] + 16);
    acc1 = __builtin_amdgcn_wmma_f32_16x16x32_f16(
        false, a[j], false, u.v, (short)0, acc1, false, false);
  }

  // ---- epilogue: scale by 1/tv (row-0 valid count + 1, matches ref), bias ----
  int tv = 1;
#pragma unroll
  for (int k = 0; k < KN; ++k) tv += (nbrs[k] >= 0) ? 1 : 0;
  const float inv = 1.0f / (float)tv;

  const int rhalf = hi * 8;  // D VGPR g: lanes 0-15 -> row g, lanes 16-31 -> row g+8
  const size_t outBase = ((size_t)b * COUT + wave * 16) * NPTS + n0 + col;
#pragma unroll
  for (int g = 0; g < 8; ++g) {
    int row = g + rhalf;
    float bv = bias[wave * 16 + row];
    out[outBase + (size_t)row * NPTS]      = acc0[g] * inv + bv;
    out[outBase + (size_t)row * NPTS + 16] = acc1[g] * inv + bv;
  }
}

// ---------------------------------------------------------------------------
extern "C" void kernel_launch(void* const* d_in, const int* in_sizes, int n_in,
                              void* d_out, int out_size, void* d_ws, size_t ws_size,
                              hipStream_t stream) {
  const float* x    = (const float*)d_in[0];  // (B, C_IN, N)
  const int*   nbrs = (const int*)d_in[1];    // (N, K)
  const float* wc   = (const float*)d_in[2];  // (C_OUT, C_IN)
  const float* wn   = (const float*)d_in[3];  // (C_OUT, C_IN, K)
  const float* bias = (const float*)d_in[4];  // (C_OUT,)
  float* out = (float*)d_out;                 // (B, C_OUT, N)

  f16* wp = (f16*)d_ws;
  f16* xt = (f16*)((char*)d_ws + XT_OFF);

  {
    int total = COUT * NCHUNK * CIN;          // 14336
    pack_weights<<<(total + 255) / 256, 256, 0, stream>>>(wc, wn, wp);
  }
  {
    int total = B_ * ROWS;                    // 640032 (incl. zero rows)
    transpose_x<<<(total + 255) / 256, 256, 0, stream>>>(x, xt);
  }
  {
    dim3 grid(B_ * NT32);                     // 20000 blocks of (b, 32 nodes)
    convhex_wmma<<<grid, 128, 0, stream>>>(xt, wp, nbrs, bias, out);
  }
}
